// ExpertGather_60885456388860
// MI455X (gfx1250) — matrix operand order repaired and checked by reading, not compiled, not run on hardware
//
#include <hip/hip_runtime.h>

// ---- WMMA / vector types (CDNA5, wave32) ----
typedef __attribute__((ext_vector_type(16))) __bf16 v16bf;
typedef __attribute__((ext_vector_type(8)))  __bf16 bf16x8;
typedef __attribute__((ext_vector_type(4)))  __bf16 bf16x4;
typedef __attribute__((ext_vector_type(4)))  float  f32x4;
typedef __attribute__((ext_vector_type(8)))  float  v8f;

#define B_  4
#define T_  4096
#define I_  1024
#define E_  16
#define K_  512
#define J_  1024

#define BM  128          // M tile (gathered rows)
#define BN  128          // N tile (J)
#define BK  32           // K step = one bf16 WMMA K
#define NS  (I_ / BK)    // 32 K-stages
#define PAD 40           // bf16 row stride in LDS (80B): 16B-aligned, conflict-free

union AB16 { v16bf v; bf16x8 h[2]; };

__global__ __launch_bounds__(256)
void moe_gather_gemm_bf16(const float* __restrict__ X,
                          const long long* __restrict__ ind,
                          const float* __restrict__ W,
                          float* __restrict__ Y)
{
    __shared__ __bf16 sA[2][BM * PAD];   // gathered X tile, [row][k], bf16, double-buffered
    __shared__ __bf16 sB[2][BN * PAD];   // W tile TRANSPOSED, [n][k], bf16, double-buffered
    __shared__ int    sIdx[BM];

    const int bx = blockIdx.x;
    const int nt = bx & 7;            // J tile      (8)
    const int mt = (bx >> 3) & 3;     // K-row tile  (4)
    const int e  = (bx >> 5) & 15;    // expert      (16)
    const int b  = bx >> 9;           // batch       (4)

    const int tid  = threadIdx.x;
    const int lane = tid & 31;
    const int wave = tid >> 5;        // 0..7
    const int wm   = wave >> 1;       // 0..3 -> 32 rows each
    const int wn   = wave & 1;        // 0..1 -> 64 cols each
    const int l16  = lane & 15;
    const int hilo = lane >> 4;       // 0 or 1

    // Cache the 128 gather indices for this block (int64 -> int32, values < T)
    if (tid < BM) {
        long long t = ind[((size_t)(b * E_ + e)) * K_ + (size_t)(mt * BM + tid)];
        sIdx[tid] = (int)t;
    }
    __syncthreads();

    // ---- staging assignments ----
    // A: 128 rows x 32 f32 per stage; thread -> (row0 + it*32, float4 seg)
    const int a_seg  = tid & 7;       // float4 within the 32-float row chunk
    const int a_row0 = tid >> 3;      // 0..31
    // B: 32 k-rows x 128 f32; thread owns a 4(k) x 4(n) micro-tile
    const int b_nseg = tid & 31;      // n = b_nseg*4
    const int b_krq  = tid >> 5;      // kr = b_krq*4

    // Hoist the gathered row base pointers out of the K loop
    const float* Xbase = X + ((size_t)b * T_) * I_;
    const float* xrow[4];
    #pragma unroll
    for (int it = 0; it < 4; ++it)
        xrow[it] = Xbase + (size_t)sIdx[a_row0 + it * 32] * I_ + a_seg * 4;

    const float* Wbase = W + ((size_t)e * I_) * J_ + (size_t)(nt * BN) + b_nseg * 4;

    v8f acc[2][4];
    #pragma unroll
    for (int t = 0; t < 2; ++t)
        #pragma unroll
        for (int u = 0; u < 4; ++u) { v8f z = {}; acc[t][u] = z; }

    // ---- register staging + convert/store helpers (inlined manually) ----
    f32x4 ar[4], br[4];

    // prologue: fetch stage 0
    #pragma unroll
    for (int it = 0; it < 4; ++it)
        ar[it] = *(const f32x4*)(xrow[it] + 0);
    #pragma unroll
    for (int r = 0; r < 4; ++r)
        br[r] = *(const f32x4*)(Wbase + (size_t)(b_krq * 4 + r) * J_);

    {   // convert + store stage 0 into buffer 0
        #pragma unroll
        for (int it = 0; it < 4; ++it) {
            bf16x4 hv = __builtin_convertvector(ar[it], bf16x4);
            *(bf16x4*)(&sA[0][(a_row0 + it * 32) * PAD + a_seg * 4]) = hv;
        }
        bf16x4 hb[4];
        #pragma unroll
        for (int r = 0; r < 4; ++r) hb[r] = __builtin_convertvector(br[r], bf16x4);
        #pragma unroll
        for (int i = 0; i < 4; ++i) {               // 4x4 register transpose
            bf16x4 kq = { hb[0][i], hb[1][i], hb[2][i], hb[3][i] };
            *(bf16x4*)(&sB[0][(b_nseg * 4 + i) * PAD + b_krq * 4]) = kq;
        }
    }
    __syncthreads();

    for (int s = 0; s < NS; ++s) {
        const int p = s & 1;
        const int chunk_next = (s + 1) * BK;

        // ---- issue next stage's global loads first (latency hidden by WMMAs) ----
        if (s + 1 < NS) {
            #pragma unroll
            for (int it = 0; it < 4; ++it)
                ar[it] = *(const f32x4*)(xrow[it] + chunk_next);
            #pragma unroll
            for (int r = 0; r < 4; ++r)
                br[r] = *(const f32x4*)(Wbase + (size_t)(chunk_next + b_krq * 4 + r) * J_);
        }

        // ---- fragments per ISA 16-bit WMMA VGPR layouts ----
        AB16 afrag[2], bfrag[4];
        #pragma unroll
        for (int t = 0; t < 2; ++t) {
            const int row = wm * 32 + t * 16 + l16;
            const int kb  = hilo * 8;    // lanes 0-15: K 0..7 & 16..23 ; lanes 16-31: +8
            afrag[t].h[0] = *(const bf16x8*)(&sA[p][row * PAD + kb]);
            afrag[t].h[1] = *(const bf16x8*)(&sA[p][row * PAD + kb + 16]);
        }
        #pragma unroll
        for (int u = 0; u < 4; ++u) {
            const int col = wn * 64 + u * 16 + l16;
            const int kb  = hilo * 8;    // lanes 0-15: K 0..7 & 16..23 ; lanes 16-31: +8
            bfrag[u].h[0] = *(const bf16x8*)(&sB[p][col * PAD + kb]);
            bfrag[u].h[1] = *(const bf16x8*)(&sB[p][col * PAD + kb + 16]);
        }

        // ---- 8 WMMAs per wave per stage ----
        #pragma unroll
        for (int t = 0; t < 2; ++t)
            #pragma unroll
            for (int u = 0; u < 4; ++u)
                acc[t][u] = __builtin_amdgcn_wmma_f32_16x16x32_bf16(
                    false, afrag[t].v, false, bfrag[u].v,
                    (short)0, acc[t][u], false, false);

        // ---- convert + store next stage into the other buffer ----
        if (s + 1 < NS) {
            const int q = p ^ 1;
            #pragma unroll
            for (int it = 0; it < 4; ++it) {
                bf16x4 hv = __builtin_convertvector(ar[it], bf16x4);
                *(bf16x4*)(&sA[q][(a_row0 + it * 32) * PAD + a_seg * 4]) = hv;
            }
            bf16x4 hb[4];
            #pragma unroll
            for (int r = 0; r < 4; ++r) hb[r] = __builtin_convertvector(br[r], bf16x4);
            #pragma unroll
            for (int i = 0; i < 4; ++i) {
                bf16x4 kq = { hb[0][i], hb[1][i], hb[2][i], hb[3][i] };
                *(bf16x4*)(&sB[q][(b_nseg * 4 + i) * PAD + b_krq * 4]) = kq;
            }
        }
        __syncthreads();
    }

    // ---- epilogue: C/D layout -> Y[b,e,m,j] (f32) ----
    #pragma unroll
    for (int t = 0; t < 2; ++t) {
        #pragma unroll
        for (int u = 0; u < 4; ++u) {
            const int m0 = mt * BM + wm * 32 + t * 16 + hilo * 8;
            const int n0 = nt * BN + wn * 64 + u * 16 + l16;
            float* outp = Y + (((size_t)(b * E_ + e) * K_ + m0) * J_ + n0);
            #pragma unroll
            for (int r = 0; r < 8; ++r)
                outp[(size_t)r * J_] = acc[t][u][r];
        }
    }
}

extern "C" void kernel_launch(void* const* d_in, const int* in_sizes, int n_in,
                              void* d_out, int out_size, void* d_ws, size_t ws_size,
                              hipStream_t stream) {
    const float*     X   = (const float*)d_in[0];
    const long long* ind = (const long long*)d_in[1];   // int64 indices
    const float*     W   = (const float*)d_in[2];
    float*           Y   = (float*)d_out;

    const int grid = B_ * E_ * (K_ / BM) * (J_ / BN);   // 4*16*4*8 = 2048
    moe_gather_gemm_bf16<<<grid, 256, 0, stream>>>(X, ind, W, Y);
}
